// MultiBranchGraphMamba_24584392802972
// MI455X (gfx1250) — matrix operand order repaired and checked by reading, not compiled, not run on hardware
//
#include <hip/hip_runtime.h>
#include <hip/hip_bf16.h>
#include <math.h>

typedef __attribute__((ext_vector_type(16))) __bf16 v16bf;
typedef __attribute__((ext_vector_type(8)))  float  v8f;

#define BB    32
#define TSEQ  128
#define NNODE 118
#define NEDGE 360
#define DH    128
#define H4    512
#define NGR   4096   /* B*T */

__device__ __forceinline__ float sigf(float x) { return 1.0f / (1.0f + __expf(-x)); }
__device__ __forceinline__ float siluf(float x) { return x * sigf(x); }
__device__ __forceinline__ float softplusf(float x) {
  return (x > 0.0f) ? (x + log1pf(__expf(-x))) : log1pf(__expf(x));
}

// ---------------------------------------------------------------------------
// Graph prep: degrees, edge norms, self norms, column-sum weights (once)
// ---------------------------------------------------------------------------
__global__ __launch_bounds__(128) void prep_kernel(const int* __restrict__ edge,
                                                   float* __restrict__ enorm,
                                                   float* __restrict__ selfn,
                                                   float* __restrict__ cw) {
  __shared__ float deg[NNODE];
  __shared__ float dinv[NNODE];
  __shared__ float cwl[NNODE];
  const int tid = threadIdx.x;
  if (tid < NNODE) deg[tid] = 1.0f;  // self loop
  __syncthreads();
  for (int e = tid; e < NEDGE; e += 128) atomicAdd(&deg[edge[NEDGE + e]], 1.0f);
  __syncthreads();
  if (tid < NNODE) {
    float di = rsqrtf(deg[tid]);
    dinv[tid] = di;
    cwl[tid] = di * di;  // self-loop contribution to column sum
  }
  __syncthreads();
  for (int e = tid; e < NEDGE; e += 128) {
    int s = edge[e], d = edge[NEDGE + e];
    float en = dinv[s] * dinv[d];
    enorm[e] = en;
    atomicAdd(&cwl[s], en);
  }
  __syncthreads();
  if (tid < NNODE) {
    selfn[tid] = dinv[tid] * dinv[tid];
    cw[tid] = cwl[tid] * (1.0f / (float)NNODE);  // fold mean-pool 1/N
  }
}

// ---------------------------------------------------------------------------
// Weight conversion to bf16 (plain and 128x128 transpose for gcn_w2)
// ---------------------------------------------------------------------------
__global__ void cvt_kernel(const float* __restrict__ s, __bf16* __restrict__ d, int n) {
  int i = blockIdx.x * blockDim.x + threadIdx.x;
  if (i < n) d[i] = (__bf16)s[i];
}
__global__ void cvt_t_kernel(const float* __restrict__ s, __bf16* __restrict__ d) {
  int i = blockIdx.x * blockDim.x + threadIdx.x;  // 16384 threads
  int n = i >> 7, k = i & 127;
  d[n * DH + k] = (__bf16)s[k * DH + n];
}

// ---------------------------------------------------------------------------
// Per-graph fused: normalize -> F=4 aggregation -> (agg@w1+b1) relu ->
// cw-weighted pool. One block per graph.
// ---------------------------------------------------------------------------
__global__ __launch_bounds__(128) void gcn_pool_kernel(
    const float* __restrict__ x, const int* __restrict__ edge,
    const float* __restrict__ mu, const float* __restrict__ sigma,
    const float* __restrict__ w1, const float* __restrict__ b1,
    const float* __restrict__ enorm, const float* __restrict__ selfn,
    const float* __restrict__ cw, float* __restrict__ pooled) {
  __shared__ float xn[NNODE * 4];
  __shared__ float agg[NNODE * 4];
  __shared__ float cwl[NNODE];
  const int tid = threadIdx.x;
  const int g = blockIdx.x;
  const size_t base = (size_t)g * NNODE * 4;
  for (int i = tid; i < NNODE * 4; i += 128) {
    int f = i & 3;
    float v = (x[base + i] - mu[f]) / sigma[f];
    xn[i] = v;
    agg[i] = selfn[i >> 2] * v;  // self-loop term
  }
  if (tid < NNODE) cwl[tid] = cw[tid];
  __syncthreads();
  for (int e = tid; e < NEDGE; e += 128) {
    int s = edge[e], d = edge[NEDGE + e];
    float w = enorm[e];
    atomicAdd(&agg[d * 4 + 0], w * xn[s * 4 + 0]);
    atomicAdd(&agg[d * 4 + 1], w * xn[s * 4 + 1]);
    atomicAdd(&agg[d * 4 + 2], w * xn[s * 4 + 2]);
    atomicAdd(&agg[d * 4 + 3], w * xn[s * 4 + 3]);
  }
  __syncthreads();
  // thread d: h1[n,d] = relu(agg[n,:]@w1[:,d]+b1[d]); pooled[d]=sum cw[n]*h1
  const float c0 = w1[tid], c1 = w1[DH + tid], c2 = w1[2 * DH + tid], c3 = w1[3 * DH + tid];
  const float bb = b1[tid];
  float acc = 0.0f;
  for (int n = 0; n < NNODE; ++n) {
    float v = agg[n * 4 + 0] * c0 + agg[n * 4 + 1] * c1 +
              agg[n * 4 + 2] * c2 + agg[n * 4 + 3] * c3 + bb;
    v = fmaxf(v, 0.0f);
    acc += cwl[n] * v;
  }
  pooled[(size_t)g * DH + tid] = acc;
}

// ---------------------------------------------------------------------------
// WMMA GEMM: C[M x N] = A[M x 128](f32) * B^T (Bbf[n*128+k], bf16) + bias1(+bias2)
// One 16x16 tile per wave, 8 waves/block, grid = (M/16, N/128).
// ---------------------------------------------------------------------------
__global__ __launch_bounds__(256) void gemm_bf16_kernel(
    const float* __restrict__ A, const __bf16* __restrict__ Bbf,
    const float* __restrict__ bias1, const float* __restrict__ bias2,
    float* __restrict__ C, int N) {
  const int tid = threadIdx.x;
  const int wave = tid >> 5, lane = tid & 31;
  const int lnm = lane & 15, hi = lane >> 4;
  const int m0 = blockIdx.x * 16;
  const int n0 = (blockIdx.y * 8 + wave) * 16;
  const int arow = m0 + lnm;
  v8f acc;
#pragma unroll
  for (int v = 0; v < 8; ++v) acc[v] = 0.0f;
#pragma unroll
  for (int ks = 0; ks < 4; ++ks) {
    const int k0 = ks * 32;
    const float4* ap = (const float4*)(A + (size_t)arow * DH + k0 + hi * 16);
    float4 q0 = ap[0], q1 = ap[1], q2 = ap[2], q3 = ap[3];
    v16bf a;
    a[0] = (__bf16)q0.x; a[1] = (__bf16)q0.y; a[2] = (__bf16)q0.z; a[3] = (__bf16)q0.w;
    a[4] = (__bf16)q1.x; a[5] = (__bf16)q1.y; a[6] = (__bf16)q1.z; a[7] = (__bf16)q1.w;
    a[8]  = (__bf16)q2.x; a[9]  = (__bf16)q2.y; a[10] = (__bf16)q2.z; a[11] = (__bf16)q2.w;
    a[12] = (__bf16)q3.x; a[13] = (__bf16)q3.y; a[14] = (__bf16)q3.z; a[15] = (__bf16)q3.w;
    const v16bf b = *(const v16bf*)(Bbf + (size_t)(n0 + lnm) * DH + k0 + hi * 16);
    acc = __builtin_amdgcn_wmma_f32_16x16x32_bf16(false, a, false, b, (short)0, acc, false, false);
  }
  const int ncol = n0 + lnm;
  const float bv = bias1[ncol] + (bias2 ? bias2[ncol] : 0.0f);
#pragma unroll
  for (int v = 0; v < 8; ++v) {
    const int m = m0 + hi * 8 + v;
    C[(size_t)m * N + ncol] = acc[v] + bv;
  }
}

// ---------------------------------------------------------------------------
// LSTM scan: one workgroup (512 thr = 16 waves). Wave (mt,nt) owns the i/f/g/o
// gate tiles for rows m0..m0+15, h-cols nt*16..+15. Cell state c stays in the
// WMMA accumulator register lanes across all 128 steps; h cycles via LDS.
// ---------------------------------------------------------------------------
__global__ __launch_bounds__(512) void lstm_scan_kernel(
    const float* __restrict__ Xp, const __bf16* __restrict__ Wbf,
    float* __restrict__ Y, int store_all) {
  __shared__ float hlds[BB * DH];
  const int tid = threadIdx.x;
  const int wave = tid >> 5, lane = tid & 31;
  const int lnm = lane & 15, hi = lane >> 4;
  const int mt = wave & 1, nt = wave >> 1;
  const int m0 = mt * 16;
  const int arow = m0 + lnm;
  const int dcol = nt * 16 + lnm;
  for (int i = tid; i < BB * DH; i += 512) hlds[i] = 0.0f;
  v8f c;
#pragma unroll
  for (int v = 0; v < 8; ++v) c[v] = 0.0f;
  __syncthreads();

  for (int t = 0; t < TSEQ; ++t) {
    v8f zi, zf, zg, zo;
#pragma unroll
    for (int v = 0; v < 8; ++v) { zi[v] = 0.f; zf[v] = 0.f; zg[v] = 0.f; zo[v] = 0.f; }
#pragma unroll
    for (int ks = 0; ks < 4; ++ks) {
      const int k0 = ks * 32;
      const float4* hp = (const float4*)&hlds[arow * DH + k0 + hi * 16];
      float4 q0 = hp[0], q1 = hp[1], q2 = hp[2], q3 = hp[3];
      v16bf a;
      a[0] = (__bf16)q0.x; a[1] = (__bf16)q0.y; a[2] = (__bf16)q0.z; a[3] = (__bf16)q0.w;
      a[4] = (__bf16)q1.x; a[5] = (__bf16)q1.y; a[6] = (__bf16)q1.z; a[7] = (__bf16)q1.w;
      a[8]  = (__bf16)q2.x; a[9]  = (__bf16)q2.y; a[10] = (__bf16)q2.z; a[11] = (__bf16)q2.w;
      a[12] = (__bf16)q3.x; a[13] = (__bf16)q3.y; a[14] = (__bf16)q3.z; a[15] = (__bf16)q3.w;
      const v16bf b0 = *(const v16bf*)(Wbf + (size_t)((nt +  0) * 16 + lnm) * DH + k0 + hi * 16);
      const v16bf b1 = *(const v16bf*)(Wbf + (size_t)((nt +  8) * 16 + lnm) * DH + k0 + hi * 16);
      const v16bf b2 = *(const v16bf*)(Wbf + (size_t)((nt + 16) * 16 + lnm) * DH + k0 + hi * 16);
      const v16bf b3 = *(const v16bf*)(Wbf + (size_t)((nt + 24) * 16 + lnm) * DH + k0 + hi * 16);
      zi = __builtin_amdgcn_wmma_f32_16x16x32_bf16(false, a, false, b0, (short)0, zi, false, false);
      zf = __builtin_amdgcn_wmma_f32_16x16x32_bf16(false, a, false, b1, (short)0, zf, false, false);
      zg = __builtin_amdgcn_wmma_f32_16x16x32_bf16(false, a, false, b2, (short)0, zg, false, false);
      zo = __builtin_amdgcn_wmma_f32_16x16x32_bf16(false, a, false, b3, (short)0, zo, false, false);
    }
    __syncthreads();  // all reads of old h complete
#pragma unroll
    for (int v = 0; v < 8; ++v) {
      const int m = m0 + hi * 8 + v;
      const float* xr = Xp + (size_t)(m * TSEQ + t) * H4;
      const float vi = zi[v] + xr[  0 + nt * 16 + lnm];
      const float vf = zf[v] + xr[128 + nt * 16 + lnm];
      const float vg = zg[v] + xr[256 + nt * 16 + lnm];
      const float vo = zo[v] + xr[384 + nt * 16 + lnm];
      const float cv = sigf(vf) * c[v] + sigf(vi) * tanhf(vg);
      c[v] = cv;
      const float hv = sigf(vo) * tanhf(cv);
      hlds[m * DH + dcol] = hv;
      if (store_all) Y[(size_t)(m * TSEQ + t) * DH + dcol] = hv;
      else if (t == TSEQ - 1) Y[(size_t)m * DH + dcol] = hv;
    }
    __syncthreads();  // new h visible before next step reads
  }
}

// ---------------------------------------------------------------------------
// Branch-selected MLP head: 128 -> 128 (silu) -> 64 (silu) -> 2, softplus+1e-6
// ---------------------------------------------------------------------------
__global__ __launch_bounds__(128) void head_kernel(
    const float* __restrict__ finalh, const int* __restrict__ bidx,
    const float* __restrict__ bemb,
    const float* __restrict__ w1, const float* __restrict__ b1,
    const float* __restrict__ w2, const float* __restrict__ b2,
    const float* __restrict__ w3, const float* __restrict__ b3,
    float* __restrict__ out) {
  __shared__ float comb[128];
  __shared__ float z1[128];
  __shared__ float z2[64];
  const int tid = threadIdx.x;
  const int b = blockIdx.x;
  const int br = bidx[b];
  comb[tid] = finalh[(size_t)b * DH + tid] + bemb[(size_t)br * DH + tid];
  __syncthreads();
  {
    const float* wr = w1 + ((size_t)br * 128 + tid) * 128;
    float acc = b1[br * 128 + tid];
    for (int i = 0; i < 128; ++i) acc += comb[i] * wr[i];
    z1[tid] = siluf(acc);
  }
  __syncthreads();
  if (tid < 64) {
    const float* wr = w2 + ((size_t)br * 64 + tid) * 128;
    float acc = b2[br * 64 + tid];
    for (int i = 0; i < 128; ++i) acc += z1[i] * wr[i];
    z2[tid] = siluf(acc);
  }
  __syncthreads();
  if (tid < 2) {
    const float* wr = w3 + ((size_t)br * 2 + tid) * 64;
    float acc = b3[br * 2 + tid];
    for (int i = 0; i < 64; ++i) acc += z2[i] * wr[i];
    out[b * 2 + tid] = softplusf(acc) + 1e-06f;
  }
}

// ---------------------------------------------------------------------------
extern "C" void kernel_launch(void* const* d_in, const int* in_sizes, int n_in,
                              void* d_out, int out_size, void* d_ws, size_t ws_size,
                              hipStream_t stream) {
  (void)in_sizes; (void)n_in; (void)out_size; (void)ws_size;
  const float* x    = (const float*)d_in[0];
  const int*   edge = (const int*)d_in[1];
  const int*   bidx = (const int*)d_in[2];
  const float* mu   = (const float*)d_in[3];
  const float* sg   = (const float*)d_in[4];
  const float* gw1  = (const float*)d_in[5];
  const float* gb1  = (const float*)d_in[6];
  const float* gw2  = (const float*)d_in[7];
  const float* gb2  = (const float*)d_in[8];
  const float* wih0 = (const float*)d_in[9];
  const float* whh0 = (const float*)d_in[10];
  const float* bih0 = (const float*)d_in[11];
  const float* bhh0 = (const float*)d_in[12];
  const float* wih1 = (const float*)d_in[13];
  const float* whh1 = (const float*)d_in[14];
  const float* bih1 = (const float*)d_in[15];
  const float* bhh1 = (const float*)d_in[16];
  const float* hw1  = (const float*)d_in[17];
  const float* hb1  = (const float*)d_in[18];
  const float* hw2  = (const float*)d_in[19];
  const float* hb2  = (const float*)d_in[20];
  const float* hw3  = (const float*)d_in[21];
  const float* hb3  = (const float*)d_in[22];
  const float* bemb = (const float*)d_in[23];
  float* out = (float*)d_out;

  char* ws = (char*)d_ws;
  size_t off = 0;
  auto take = [&](size_t bytes) -> char* {
    char* p = ws + off;
    off = (off + bytes + 255) & ~(size_t)255;
    return p;
  };
  float*  enorm  = (float*)take(NEDGE * 4);
  float*  selfn  = (float*)take(NNODE * 4);
  float*  cw     = (float*)take(NNODE * 4);
  float*  P      = (float*)take((size_t)NGR * DH * 4);   // pooled, reused as ys0
  float*  Q      = (float*)take((size_t)NGR * DH * 4);   // emb
  float*  X      = (float*)take((size_t)NGR * H4 * 4);   // x-projections (reused)
  float*  Fh     = (float*)take((size_t)BB * DH * 4);    // final hidden
  __bf16* w2t_b  = (__bf16*)take((size_t)DH * DH * 2);
  __bf16* wih0_b = (__bf16*)take((size_t)H4 * DH * 2);
  __bf16* whh0_b = (__bf16*)take((size_t)H4 * DH * 2);
  __bf16* wih1_b = (__bf16*)take((size_t)H4 * DH * 2);
  __bf16* whh1_b = (__bf16*)take((size_t)H4 * DH * 2);

  prep_kernel<<<1, 128, 0, stream>>>(edge, enorm, selfn, cw);
  cvt_kernel<<<256, 256, 0, stream>>>(wih0, wih0_b, H4 * DH);
  cvt_kernel<<<256, 256, 0, stream>>>(whh0, whh0_b, H4 * DH);
  cvt_kernel<<<256, 256, 0, stream>>>(wih1, wih1_b, H4 * DH);
  cvt_kernel<<<256, 256, 0, stream>>>(whh1, whh1_b, H4 * DH);
  cvt_t_kernel<<<64, 256, 0, stream>>>(gw2, w2t_b);

  gcn_pool_kernel<<<NGR, 128, 0, stream>>>(x, edge, mu, sg, gw1, gb1,
                                           enorm, selfn, cw, P);
  // emb = pooled @ w2 + b2
  gemm_bf16_kernel<<<dim3(NGR / 16, 1), 256, 0, stream>>>(P, w2t_b, gb2, nullptr, Q, DH);
  // layer-0 input projection for all t
  gemm_bf16_kernel<<<dim3(NGR / 16, 4), 256, 0, stream>>>(Q, wih0_b, bih0, bhh0, X, H4);
  // layer-0 recurrent scan -> ys0 (in P)
  lstm_scan_kernel<<<1, 512, 0, stream>>>(X, whh0_b, P, 1);
  // layer-1 input projection for all t
  gemm_bf16_kernel<<<dim3(NGR / 16, 4), 256, 0, stream>>>(P, wih1_b, bih1, bhh1, X, H4);
  // layer-1 recurrent scan -> final hidden only
  lstm_scan_kernel<<<1, 512, 0, stream>>>(X, whh1_b, Fh, 0);

  head_kernel<<<BB, 128, 0, stream>>>(Fh, bidx, bemb, hw1, hb1, hw2, hb2, hw3, hb3, out);
}
